// CausalLM_7138235646701
// MI455X (gfx1250) — matrix operand order repaired and checked by reading, not compiled, not run on hardware
//
#include <hip/hip_runtime.h>
#include <hip/hip_bf16.h>
#include <stdint.h>

typedef __attribute__((ext_vector_type(16))) __bf16 v16bf;
typedef __attribute__((ext_vector_type(8)))  float  v8f;

union U16 { v16bf v; __bf16 e[16]; unsigned u[8]; };

// A-matrix (16x32 bf16) per-lane K index for VGPR r, lane-half h (ISA 7.12.2)
__device__ __forceinline__ int basekA(int r, int h) { return 2*r + (r >= 4 ? 8 : 0) + 8*h; }
// B-matrix (32x16 bf16) per-lane K index for VGPR r, lane-half h
__device__ __forceinline__ int basekB(int r, int h) { return 2*r + (r >= 4 ? 8 : 0) + 8*h; }

#define Bn 2
#define Hn 16
#define Sn 2048
#define Dn 128
#define LOG2E 1.4426950408889634f
#define MASKV (-3.0e38f)

// butterfly reduce over 16 lanes via ds_swizzle (and_mask=0x1f, xor in offset[14:10])
#define SWZ_MAXF(x, imm) fmaxf((x), __uint_as_float(__builtin_amdgcn_ds_swizzle(__float_as_uint(x), (imm))))
#define SWZ_ADDF(x, imm) ((x) + __uint_as_float(__builtin_amdgcn_ds_swizzle(__float_as_uint(x), (imm))))

// ---------------- pre-pass: K -> bf16 [B,H,S,D]; V*ctx -> bf16 transposed [B,H,D,S] ----------------
__global__ __launch_bounds__(256) void prep_kv(
    const float* __restrict__ K, const float* __restrict__ V,
    const float* __restrict__ CM, __bf16* __restrict__ Kw, __bf16* __restrict__ Vw)
{
  __shared__ __bf16 tile[128][66];   // [d][s_local], padded (odd dword stride)
  const int tid = threadIdx.x;
  const int b = blockIdx.z, h = blockIdx.y;
  const int s0 = blockIdx.x * 64;
  const size_t bh = (size_t)(b * Hn + h) * Sn;

  #pragma unroll
  for (int it = 0; it < 8; ++it) {
    int idx = tid + it * 256;              // 2048 float4 slots = 64 rows x 32
    int row = idx >> 5;
    int c4  = (idx & 31) << 2;
    const float4 kk = *(const float4*)(K + (bh + s0 + row) * Dn + c4);
    __bf16* kd = Kw + (bh + s0 + row) * Dn + c4;
    kd[0] = (__bf16)kk.x; kd[1] = (__bf16)kk.y; kd[2] = (__bf16)kk.z; kd[3] = (__bf16)kk.w;
    const float cm = CM[(size_t)b * Sn + s0 + row];
    const float4 vv = *(const float4*)(V + (bh + s0 + row) * Dn + c4);
    tile[c4+0][row] = (__bf16)(vv.x * cm); tile[c4+1][row] = (__bf16)(vv.y * cm);
    tile[c4+2][row] = (__bf16)(vv.z * cm); tile[c4+3][row] = (__bf16)(vv.w * cm);
  }
  __syncthreads();
  #pragma unroll
  for (int it = 0; it < 4; ++it) {
    int idx = tid + it * 256;              // 1024 chunks of 8 bf16 = 128 rows x 8
    int d = idx >> 3;
    int c = (idx & 7) << 3;
    __bf16* vd = Vw + ((size_t)(b * Hn + h) * Dn + d) * Sn + s0 + c;
    #pragma unroll
    for (int i = 0; i < 8; ++i) vd[i] = tile[d][c + i];
  }
}

// ---------------- main flash-attention kernel (bf16 K/V from workspace) ----------------
__global__ __launch_bounds__(256) void fa_main(
    const float* __restrict__ Q, const __bf16* __restrict__ Kw,
    const __bf16* __restrict__ Vw, const float* __restrict__ AM,
    const float* __restrict__ HM, float* __restrict__ O)
{
  // double-buffered tiles; strides are multiples of 16B for async b128 LDS writes,
  // padded dword strides (68 / 36) for conflict-free fragment reads
  __shared__ __bf16 kt[2][64][136];   // [kv][d]
  __shared__ __bf16 vt[2][128][72];   // [d][kv]
  __shared__ __bf16 pt[8][16][66];    // per-wave P scratch [m][kv]

  const int tid  = threadIdx.x;
  const int wave = tid >> 5;
  const int lane = tid & 31;
  const int half = lane >> 4;
  const int nl   = lane & 15;

  const int b = blockIdx.z, h = blockIdx.y;
  const int qBase = blockIdx.x * 128;
  const int q0 = qBase + wave * 16;

  const size_t bh = (size_t)(b * Hn + h) * Sn;

  // Q fragments (A layout), log2(e) folded in so softmax can use raw exp2
  const float* Qp = Q + (bh + q0 + nl) * Dn;
  U16 qa[4];
  #pragma unroll
  for (int c = 0; c < 4; ++c) {
    #pragma unroll
    for (int r = 0; r < 8; ++r) {
      int k = c * 32 + basekA(r, half);
      float2 f = *(const float2*)(Qp + k);
      qa[c].e[2*r]   = (__bf16)(f.x * LOG2E);
      qa[c].e[2*r+1] = (__bf16)(f.y * LOG2E);
    }
  }

  v8f acc[8];
  #pragma unroll
  for (int dt = 0; dt < 8; ++dt) acc[dt] = (v8f){0.f,0.f,0.f,0.f,0.f,0.f,0.f,0.f};
  float row_max[8], row_sum[8];
  #pragma unroll
  for (int v = 0; v < 8; ++v) { row_max[v] = MASKV; row_sum[v] = 0.f; }

  const unsigned kt_base = (unsigned)(uintptr_t)&kt[0][0][0];
  const unsigned vt_base = (unsigned)(uintptr_t)&vt[0][0][0];
  const uint64_t kg_base = (uint64_t)(uintptr_t)(Kw + bh * Dn);
  const uint64_t vg_base = (uint64_t)(uintptr_t)(Vw + (size_t)(b * Hn + h) * Dn * Sn);

  // issue one 64x128 bf16 K tile + 128x64 bf16 V tile as async global->LDS b128 copies
  auto issue_tile = [&](int kv0, int bf) {
    #pragma unroll
    for (int it = 0; it < 4; ++it) {
      int idx = tid + it * 256;                     // 1024 chunks of 16B each
      int row = idx >> 4;                           // K: 64 rows x 16 chunks
      int cK  = (idx & 15) << 4;                    // byte offset in row
      unsigned lk = kt_base + (unsigned)bf * (64u*136u*2u) + (unsigned)row * (136u*2u) + (unsigned)cK;
      uint64_t gk = kg_base + ((uint64_t)(kv0 + row) * (Dn*2u)) + (uint64_t)cK;
      asm volatile("global_load_async_to_lds_b128 %0, %1, off" :: "v"(lk), "v"(gk) : "memory");
      int d  = idx >> 3;                            // V: 128 rows x 8 chunks
      int cV = (idx & 7) << 4;
      unsigned lv = vt_base + (unsigned)bf * (128u*72u*2u) + (unsigned)d * (72u*2u) + (unsigned)cV;
      uint64_t gv = vg_base + ((uint64_t)d * (Sn*2u)) + (uint64_t)(kv0*2) + (uint64_t)cV;
      asm volatile("global_load_async_to_lds_b128 %0, %1, off" :: "v"(lv), "v"(gv) : "memory");
    }
  };

  const int njt = qBase / 64 + 2;                   // causal tile count (kv tiles of 64)
  issue_tile(0, 0);

  for (int j = 0; j < njt; ++j) {
    const int kv0 = j * 64;
    const int bf = j & 1;
    if (j + 1 < njt) {
      issue_tile(kv0 + 64, bf ^ 1);                 // prefetch next tile into other buffer
      asm volatile("s_wait_asynccnt 0x8" ::: "memory");  // in-order: current tile's 8 done
    } else {
      asm volatile("s_wait_asynccnt 0x0" ::: "memory");
    }
    __syncthreads();

    if (kv0 <= q0 + 15) {                           // wave-uniform causal tile skip
      // ---- S = Q K^T : 16x64 scores as four 16x16 tiles ----
      v8f s[4];
      #pragma unroll
      for (int t = 0; t < 4; ++t) {
        s[t] = (v8f){0.f,0.f,0.f,0.f,0.f,0.f,0.f,0.f};
        #pragma unroll
        for (int c = 0; c < 4; ++c) {
          U16 kb;
          #pragma unroll
          for (int r = 0; r < 8; ++r)
            kb.u[r] = *(const unsigned*)&kt[bf][t*16 + nl][c*32 + basekB(r, half)];
          s[t] = __builtin_amdgcn_wmma_f32_16x16x32_bf16(false, qa[c].v, false, kb.v, (short)0, s[t], false, false);
        }
      }

      float amS[4];
      #pragma unroll
      for (int t = 0; t < 4; ++t)
        amS[t] = AM[(size_t)b * Sn + kv0 + t*16 + nl] * LOG2E;

      const bool edge = (kv0 + 63 > q0);            // only the diagonal tile needs masking

      #pragma unroll
      for (int v = 0; v < 8; ++v) {
        const int qrow = q0 + v + 8*half;
        float x[4];
        #pragma unroll
        for (int t = 0; t < 4; ++t) x[t] = s[t][v] + amS[t];
        if (edge) {
          #pragma unroll
          for (int t = 0; t < 4; ++t)
            if (kv0 + t*16 + nl > qrow) x[t] = MASKV;
        }
        float tm = fmaxf(fmaxf(x[0], x[1]), fmaxf(x[2], x[3]));
        tm = SWZ_MAXF(tm, 0x041F); tm = SWZ_MAXF(tm, 0x081F);
        tm = SWZ_MAXF(tm, 0x101F); tm = SWZ_MAXF(tm, 0x201F);
        const float nm = fmaxf(row_max[v], tm);
        const float alpha = __builtin_amdgcn_exp2f(row_max[v] - nm);
        row_max[v] = nm;
        float p[4];
        #pragma unroll
        for (int t = 0; t < 4; ++t) p[t] = __builtin_amdgcn_exp2f(x[t] - nm);
        float ps = (p[0] + p[1]) + (p[2] + p[3]);
        ps = SWZ_ADDF(ps, 0x041F); ps = SWZ_ADDF(ps, 0x081F);
        ps = SWZ_ADDF(ps, 0x101F); ps = SWZ_ADDF(ps, 0x201F);
        row_sum[v] = row_sum[v] * alpha + ps;
        #pragma unroll
        for (int dt = 0; dt < 8; ++dt) acc[dt][v] *= alpha;
        #pragma unroll
        for (int t = 0; t < 4; ++t)
          pt[wave][v + 8*half][t*16 + nl] = (__bf16)p[t];  // ctx already folded into V
      }

      asm volatile("s_wait_dscnt 0x0" ::: "memory");  // wave-local P bounce

      // ---- O += P V : A = P (16x64 -> 2 chunks), B = V tiles ----
      U16 pa[2];
      #pragma unroll
      for (int kc = 0; kc < 2; ++kc)
        #pragma unroll
        for (int r = 0; r < 8; ++r)
          pa[kc].u[r] = *(const unsigned*)&pt[wave][nl][kc*32 + basekA(r, half)];

      #pragma unroll
      for (int dt = 0; dt < 8; ++dt) {
        #pragma unroll
        for (int kc = 0; kc < 2; ++kc) {
          U16 vb;
          #pragma unroll
          for (int r = 0; r < 8; ++r)
            vb.u[r] = *(const unsigned*)&vt[bf][dt*16 + nl][kc*32 + basekB(r, half)];
          acc[dt] = __builtin_amdgcn_wmma_f32_16x16x32_bf16(false, pa[kc].v, false, vb.v, (short)0, acc[dt], false, false);
        }
      }
    }
    __syncthreads();
  }

  // ---- epilogue: O = acc * head_mask / row_sum ----
  const float hm = HM[h];
  float* Op = O + (bh + q0) * Dn;
  #pragma unroll
  for (int v = 0; v < 8; ++v) {
    const float sc = hm / row_sum[v];
    #pragma unroll
    for (int dt = 0; dt < 8; ++dt)
      Op[(size_t)(v + 8*half) * Dn + dt*16 + nl] = acc[dt][v] * sc;
  }
}

// ---------------- fallback fused kernel (round-1 version, used if ws too small) ----------------
__global__ __launch_bounds__(256) void fa_fused(
    const float* __restrict__ Q, const float* __restrict__ K,
    const float* __restrict__ V, const float* __restrict__ AM,
    const float* __restrict__ HM, const float* __restrict__ CM,
    float* __restrict__ O)
{
  __shared__ __bf16 kt[32][132];
  __shared__ __bf16 vt[128][34];
  __shared__ __bf16 pt[8][16][34];

  const int tid  = threadIdx.x;
  const int wave = tid >> 5;
  const int lane = tid & 31;
  const int half = lane >> 4;
  const int nl   = lane & 15;

  const int b = blockIdx.z, h = blockIdx.y;
  const int qBase = blockIdx.x * 128;
  const int q0 = qBase + wave * 16;

  const size_t bh = (size_t)(b * Hn + h) * Sn;
  const float* Qp = Q + (bh + q0 + nl) * Dn;
  const float* Kp = K + bh * Dn;
  const float* Vp = V + bh * Dn;

  U16 qa[4];
  #pragma unroll
  for (int c = 0; c < 4; ++c)
    #pragma unroll
    for (int r = 0; r < 8; ++r) {
      int k = c*32 + basekA(r, half);
      float2 f = *(const float2*)(Qp + k);
      qa[c].e[2*r]   = (__bf16)(f.x * LOG2E);
      qa[c].e[2*r+1] = (__bf16)(f.y * LOG2E);
    }

  v8f acc[8];
  #pragma unroll
  for (int dt = 0; dt < 8; ++dt) acc[dt] = (v8f){0.f,0.f,0.f,0.f,0.f,0.f,0.f,0.f};
  float row_max[8], row_sum[8];
  #pragma unroll
  for (int v = 0; v < 8; ++v) { row_max[v] = MASKV; row_sum[v] = 0.f; }

  const int njt = qBase/32 + 4;
  for (int j = 0; j < njt; ++j) {
    const int kv0 = j * 32;
    #pragma unroll
    for (int it = 0; it < 4; ++it) {
      int idx = tid + it * 256;
      int row = idx >> 5;
      int c4  = (idx & 31) << 2;
      float4 kk = *(const float4*)(Kp + (size_t)(kv0 + row) * Dn + c4);
      kt[row][c4+0] = (__bf16)kk.x; kt[row][c4+1] = (__bf16)kk.y;
      kt[row][c4+2] = (__bf16)kk.z; kt[row][c4+3] = (__bf16)kk.w;
      float4 vv = *(const float4*)(Vp + (size_t)(kv0 + row) * Dn + c4);
      vt[c4+0][row] = (__bf16)vv.x; vt[c4+1][row] = (__bf16)vv.y;
      vt[c4+2][row] = (__bf16)vv.z; vt[c4+3][row] = (__bf16)vv.w;
    }
    __syncthreads();

    if (kv0 <= q0 + 15) {
      v8f s0 = (v8f){0.f,0.f,0.f,0.f,0.f,0.f,0.f,0.f};
      v8f s1 = s0;
      #pragma unroll
      for (int c = 0; c < 4; ++c) {
        U16 kb0, kb1;
        #pragma unroll
        for (int r = 0; r < 8; ++r) {
          int dk = c*32 + basekB(r, half);
          kb0.u[r] = *(const unsigned*)&kt[nl][dk];
          kb1.u[r] = *(const unsigned*)&kt[16 + nl][dk];
        }
        s0 = __builtin_amdgcn_wmma_f32_16x16x32_bf16(false, qa[c].v, false, kb0.v, (short)0, s0, false, false);
        s1 = __builtin_amdgcn_wmma_f32_16x16x32_bf16(false, qa[c].v, false, kb1.v, (short)0, s1, false, false);
      }

      const float am0 = AM[(size_t)b * Sn + kv0 + nl] * LOG2E;
      const float am1 = AM[(size_t)b * Sn + kv0 + 16 + nl] * LOG2E;
      const float cm0 = CM[(size_t)b * Sn + kv0 + nl];
      const float cm1 = CM[(size_t)b * Sn + kv0 + 16 + nl];

      #pragma unroll
      for (int v = 0; v < 8; ++v) {
        const int qrow = q0 + v + 8*half;
        float x0 = ((kv0 + nl)      <= qrow ? s0[v] + am0 : MASKV);
        float x1 = ((kv0 + 16 + nl) <= qrow ? s1[v] + am1 : MASKV);
        float t = fmaxf(x0, x1);
        t = SWZ_MAXF(t, 0x041F); t = SWZ_MAXF(t, 0x081F);
        t = SWZ_MAXF(t, 0x101F); t = SWZ_MAXF(t, 0x201F);
        float nm    = fmaxf(row_max[v], t);
        float alpha = __builtin_amdgcn_exp2f(row_max[v] - nm);
        row_max[v]  = nm;
        float p0 = __builtin_amdgcn_exp2f(x0 - nm);
        float p1 = __builtin_amdgcn_exp2f(x1 - nm);
        float ps = p0 + p1;
        ps = SWZ_ADDF(ps, 0x041F); ps = SWZ_ADDF(ps, 0x081F);
        ps = SWZ_ADDF(ps, 0x101F); ps = SWZ_ADDF(ps, 0x201F);
        row_sum[v] = row_sum[v] * alpha + ps;
        #pragma unroll
        for (int dt = 0; dt < 8; ++dt) acc[dt][v] *= alpha;
        pt[wave][v + 8*half][nl]      = (__bf16)(p0 * cm0);
        pt[wave][v + 8*half][16 + nl] = (__bf16)(p1 * cm1);
      }

      asm volatile("s_wait_dscnt 0x0" ::: "memory");

      U16 pa;
      #pragma unroll
      for (int r = 0; r < 8; ++r)
        pa.u[r] = *(const unsigned*)&pt[wave][nl][basekA(r, half)];

      #pragma unroll
      for (int dt = 0; dt < 8; ++dt) {
        U16 vb;
        #pragma unroll
        for (int r = 0; r < 8; ++r)
          vb.u[r] = *(const unsigned*)&vt[dt*16 + nl][basekB(r, half)];
        acc[dt] = __builtin_amdgcn_wmma_f32_16x16x32_bf16(false, pa.v, false, vb.v, (short)0, acc[dt], false, false);
      }
    }
    __syncthreads();
  }

  const float hm = HM[h];
  float* Op = O + (bh + q0) * Dn;
  #pragma unroll
  for (int v = 0; v < 8; ++v) {
    const float sc = hm / row_sum[v];
    #pragma unroll
    for (int dt = 0; dt < 8; ++dt)
      Op[(size_t)(v + 8*half) * Dn + dt*16 + nl] = acc[dt][v] * sc;
  }
}

extern "C" void kernel_launch(void* const* d_in, const int* in_sizes, int n_in,
                              void* d_out, int out_size, void* d_ws, size_t ws_size,
                              hipStream_t stream) {
  const float* Q  = (const float*)d_in[0];
  const float* K  = (const float*)d_in[1];
  const float* V  = (const float*)d_in[2];
  const float* AM = (const float*)d_in[3];
  const float* HM = (const float*)d_in[4];
  const float* CM = (const float*)d_in[5];
  float* O = (float*)d_out;
  (void)in_sizes; (void)n_in; (void)out_size;

  const size_t kwBytes = (size_t)Bn * Hn * Sn * Dn * sizeof(__bf16);  // 16 MiB each
  if (ws_size >= 2 * kwBytes) {
    __bf16* Kw = (__bf16*)d_ws;
    __bf16* Vw = (__bf16*)((char*)d_ws + kwBytes);
    prep_kv<<<dim3(Sn / 64, Hn, Bn), dim3(256), 0, stream>>>(K, V, CM, Kw, Vw);
    fa_main<<<dim3(Sn / 128, Hn, Bn), dim3(256), 0, stream>>>(Q, Kw, Vw, AM, HM, O);
  } else {
    fa_fused<<<dim3(Sn / 128, Hn, Bn), dim3(256), 0, stream>>>(Q, K, V, AM, HM, CM, O);
  }
}
